// EdgeUpdateLayerMetapath_42425686950359
// MI455X (gfx1250) — compile-verified, matrix operand order checked by reading, not run on hardware
//
#include <hip/hip_runtime.h>
#include <hip/hip_bf16.h>

// ---------------------------------------------------------------------------
// Graph-transformer block for MI455X (gfx1250, wave32, WMMA).
// GEMMs: bf16 v_wmma_f32_16x16x32_bf16, LDS staged in fragment-friendly
// layout so every fragment is 2x ds_load_b128. 4 WMMAs per wave per K-chunk.
// Edge attention: gathers (L2-resident Q/K/V) + fp32 global atomics.
// ---------------------------------------------------------------------------

typedef __bf16 bf16;
typedef bf16  v16bf __attribute__((ext_vector_type(16)));
typedef bf16  v8bf  __attribute__((ext_vector_type(8)));
typedef float v8f   __attribute__((ext_vector_type(8)));

#define D_MODEL 512
#define NHEAD   8

// LDS row stride (bf16 units): 40 -> 80 bytes, keeps rows 16B-aligned and
// spreads banks (80B = 20 banks apart between consecutive rows).
#define LDSPITCH 40

__device__ __forceinline__ v16bf frag_from_lds(const bf16* rowp, int kbase) {
    // per-lane 16-bit fragment = two contiguous 8-element runs:
    // k in [kbase, kbase+8) and [kbase+16, kbase+24)
    v8bf lo = *(const v8bf*)(rowp + kbase);
    v8bf hi = *(const v8bf*)(rowp + kbase + 16);
    return __builtin_shufflevector(lo, hi, 0, 1, 2, 3, 4, 5, 6, 7,
                                           8, 9, 10, 11, 12, 13, 14, 15);
}

// ----------------------------- zero fill -----------------------------------
__global__ __launch_bounds__(256)
void zero_f32(float* __restrict__ p, long n) {
    long i = (long)blockIdx.x * blockDim.x + threadIdx.x;
    if (i < n) p[i] = 0.0f;
}

// ----------------------------- GEMM (bf16 WMMA) ----------------------------
// C[M,Nc] = act(A[M,K] @ W[K,Nc] + bias), fp32 in/out, bf16 compute.
// Block: 256 threads = 8 waves; 64x128 block tile; each wave a 32x32 subtile
// (2 A-frags x 2 B-frags = 4 WMMAs per 32-wide K chunk).
__global__ __launch_bounds__(256)
void gemm_bias_act(const float* __restrict__ A, const float* __restrict__ W,
                   const float* __restrict__ bias, float* __restrict__ C,
                   int M, int K, int Nc, int do_relu)
{
    __shared__ bf16 As[64][LDSPITCH];      // A tile, row-major [m][k]
    __shared__ bf16 Bt[128][LDSPITCH];     // B tile, TRANSPOSED [n][k]

    const int tid  = threadIdx.x;
    const int lane = tid & 31;
    const int wave = tid >> 5;
    const int wm = (wave >> 2) * 32;       // 0 or 32
    const int wn = (wave & 3) * 32;        // 0,32,64,96
    const int bm = blockIdx.x * 64;
    const int bn = blockIdx.y * 128;

    // A staging map: 256 threads x 8 floats = 64x32 tile
    const int alr = tid >> 2;              // 0..63 row
    const int alc = (tid & 3) * 8;         // 0,8,16,24 k
    // B staging map: 2 k-rows x 8 cols per thread (16 kpairs x 16 colgroups)
    const int bkp = (tid >> 4) * 2;        // k pair base: 0,2,..,30
    const int bnc = (tid & 15) * 8;        // col base: 0..120

    v8f acc00 = {}, acc01 = {}, acc10 = {}, acc11 = {};

    const int mrow = wm + (lane & 15);
    const int ncol = wn + (lane & 15);
    const int kbase = (lane >> 4) << 3;    // 0 or 8

    for (int k0 = 0; k0 < K; k0 += 32) {
        // --- stage A chunk: one 16B packed LDS store per thread ---
        {
            const int gr = bm + alr;
            float4 u0 = {0.f, 0.f, 0.f, 0.f}, u1 = {0.f, 0.f, 0.f, 0.f};
            if (gr < M) {
                const float4* s4 = (const float4*)(A + (size_t)gr * K + k0 + alc);
                u0 = s4[0];
                u1 = s4[1];
                if (k0 + 32 < K)
                    __builtin_prefetch(A + (size_t)gr * K + k0 + 32 + alc, 0, 1);
            }
            v8bf pk;
            pk[0] = (bf16)u0.x; pk[1] = (bf16)u0.y; pk[2] = (bf16)u0.z; pk[3] = (bf16)u0.w;
            pk[4] = (bf16)u1.x; pk[5] = (bf16)u1.y; pk[6] = (bf16)u1.z; pk[7] = (bf16)u1.w;
            *(v8bf*)&As[alr][alc] = pk;
        }
        // --- stage B chunk transposed: pack (k, k+1) pairs into b32 stores ---
        {
            const float* r0 = W + (size_t)(k0 + bkp) * Nc + bn + bnc;
            const float* r1 = r0 + Nc;
            const float4* r04 = (const float4*)r0;
            const float4* r14 = (const float4*)r1;
            float4 a0 = r04[0], a1 = r04[1];
            float4 b0 = r14[0], b1 = r14[1];
            if (k0 + 32 < K) __builtin_prefetch(r0 + (size_t)32 * Nc, 0, 1);
            float va[8] = {a0.x, a0.y, a0.z, a0.w, a1.x, a1.y, a1.z, a1.w};
            float vb[8] = {b0.x, b0.y, b0.z, b0.w, b1.x, b1.y, b1.z, b1.w};
            #pragma unroll
            for (int i = 0; i < 8; ++i) {
                // Bt[n][k] = W[k][n]; write 2 bf16 (k even) as one aligned b32
                bf16 two[2] = {(bf16)va[i], (bf16)vb[i]};
                *(unsigned int*)&Bt[bnc + i][bkp] = *(unsigned int*)two;
            }
        }
        __syncthreads();

        // --- fragments: each is 2x ds_load_b128 from a contiguous LDS row ---
        v16bf a0 = frag_from_lds(&As[mrow][0],       kbase);
        v16bf a1 = frag_from_lds(&As[mrow + 16][0],  kbase);
        v16bf b0 = frag_from_lds(&Bt[ncol][0],       kbase);
        v16bf b1 = frag_from_lds(&Bt[ncol + 16][0],  kbase);

        acc00 = __builtin_amdgcn_wmma_f32_16x16x32_bf16(false, a0, false, b0, (short)0, acc00, false, false);
        acc01 = __builtin_amdgcn_wmma_f32_16x16x32_bf16(false, a0, false, b1, (short)0, acc01, false, false);
        acc10 = __builtin_amdgcn_wmma_f32_16x16x32_bf16(false, a1, false, b0, (short)0, acc10, false, false);
        acc11 = __builtin_amdgcn_wmma_f32_16x16x32_bf16(false, a1, false, b1, (short)0, acc11, false, false);
        __syncthreads();
    }

    // --- epilogue: C/D layout: lane%16 -> N, vgpr r -> M=r (+8 for hi lanes)
    const int cn0 = bn + wn + (lane & 15);
    const int cn1 = cn0 + 16;
    const int m0  = bm + wm + ((lane >> 4) << 3);
    const float bv0 = bias[cn0];
    const float bv1 = bias[cn1];
    #pragma unroll
    for (int r = 0; r < 8; ++r) {
        const int gmA = m0 + r;
        const int gmB = gmA + 16;
        if (gmA < M) {
            float v0 = acc00[r] + bv0;
            float v1 = acc01[r] + bv1;
            if (do_relu) { v0 = fmaxf(v0, 0.f); v1 = fmaxf(v1, 0.f); }
            C[(size_t)gmA * Nc + cn0] = v0;
            C[(size_t)gmA * Nc + cn1] = v1;
        }
        if (gmB < M) {
            float v0 = acc10[r] + bv0;
            float v1 = acc11[r] + bv1;
            if (do_relu) { v0 = fmaxf(v0, 0.f); v1 = fmaxf(v1, 0.f); }
            C[(size_t)gmB * Nc + cn0] = v0;
            C[(size_t)gmB * Nc + cn1] = v1;
        }
    }
}

// ----------------------------- edge attention ------------------------------
// One wave32 per edge. Each lane covers 16 contiguous floats of the 512-wide
// row (4 lanes per head, DK=64). score = exp(clamp(k_src . q_dst / 8, +-5)).
// Scatter score -> z[dst,h] and score * v_src -> wv[dst,:] with f32 atomics.
__global__ __launch_bounds__(256)
void edge_attn(const float* __restrict__ Q, const float* __restrict__ Kd,
               const float* __restrict__ V, const int* __restrict__ esrc,
               const int* __restrict__ edst, float* __restrict__ wv,
               float* __restrict__ z, int E)
{
    const int gwarp = (int)((blockIdx.x * (unsigned)blockDim.x + threadIdx.x) >> 5);
    const int lane  = threadIdx.x & 31;
    if (gwarp >= E) return;

    const int s = esrc[gwarp];
    const int d = edst[gwarp];

    const float4* kp = (const float4*)(Kd + (size_t)s * D_MODEL);
    const float4* qp = (const float4*)(Q  + (size_t)d * D_MODEL);

    float part = 0.0f;
    #pragma unroll
    for (int i = 0; i < 4; ++i) {
        const float4 a = kp[lane * 4 + i];
        const float4 b = qp[lane * 4 + i];
        part += a.x * b.x + a.y * b.y + a.z * b.z + a.w * b.w;
    }
    // reduce over the 4 lanes of this head
    part += __shfl_xor(part, 1, 32);
    part += __shfl_xor(part, 2, 32);

    const float score = __expf(fminf(fmaxf(part * 0.125f, -5.0f), 5.0f));

    if ((lane & 3) == 0)
        atomicAdd(&z[(size_t)d * NHEAD + (lane >> 2)], score);

    const float4* vp = (const float4*)(V + (size_t)s * D_MODEL);
    float* wvp = wv + (size_t)d * D_MODEL;
    #pragma unroll
    for (int i = 0; i < 4; ++i) {
        const float4 a = vp[lane * 4 + i];
        const int off = (lane * 4 + i) * 4;
        atomicAdd(&wvp[off + 0], a.x * score);
        atomicAdd(&wvp[off + 1], a.y * score);
        atomicAdd(&wvp[off + 2], a.z * score);
        atomicAdd(&wvp[off + 3], a.w * score);
    }
}

// ----------------------------- o = wv / z (float4) -------------------------
__global__ __launch_bounds__(256)
void finalize_o(const float* __restrict__ wv, const float* __restrict__ z,
                float* __restrict__ o, int n4)
{
    const int i4 = blockIdx.x * blockDim.x + threadIdx.x;   // float4 index
    if (i4 >= n4) return;
    const int node = i4 >> 7;          // / 128 float4s per row
    const int h    = (i4 >> 4) & 7;    // / 16 float4s per head
    const float zz = z[node * NHEAD + h];
    const float inv = 1.0f / (zz == 0.0f ? 1.0f : zz);
    float4 v = ((const float4*)wv)[i4];
    v.x *= inv; v.y *= inv; v.z *= inv; v.w *= inv;
    ((float4*)o)[i4] = v;
}

// ----------------------------- residual + layernorm ------------------------
// One 128-thread block per row of 512; float4 per thread; wave reductions.
__global__ __launch_bounds__(128)
void add_layernorm(const float* __restrict__ X, const float* __restrict__ Y,
                   const float* __restrict__ gamma, const float* __restrict__ beta,
                   float* __restrict__ out)
{
    __shared__ float red[4];
    const int row = blockIdx.x;
    const int t = threadIdx.x;
    const float4 xv = ((const float4*)(X + (size_t)row * D_MODEL))[t];
    const float4 yv = ((const float4*)(Y + (size_t)row * D_MODEL))[t];

    float v[4] = {xv.x + yv.x, xv.y + yv.y, xv.z + yv.z, xv.w + yv.w};
    float s = v[0] + v[1] + v[2] + v[3];
    #pragma unroll
    for (int o = 16; o > 0; o >>= 1) s += __shfl_xor(s, o, 32);
    if ((t & 31) == 0) red[t >> 5] = s;
    __syncthreads();
    const float mean = (red[0] + red[1] + red[2] + red[3]) * (1.0f / D_MODEL);
    __syncthreads();

    float s2 = 0.0f;
    #pragma unroll
    for (int i = 0; i < 4; ++i) { const float dd = v[i] - mean; s2 += dd * dd; }
    #pragma unroll
    for (int o = 16; o > 0; o >>= 1) s2 += __shfl_xor(s2, o, 32);
    if ((t & 31) == 0) red[t >> 5] = s2;
    __syncthreads();
    const float var = (red[0] + red[1] + red[2] + red[3]) * (1.0f / D_MODEL);
    const float inv = rsqrtf(var + 1e-5f);

    const float4 gv = ((const float4*)gamma)[t];
    const float4 bv = ((const float4*)beta)[t];
    float4 ov;
    ov.x = (v[0] - mean) * inv * gv.x + bv.x;
    ov.y = (v[1] - mean) * inv * gv.y + bv.y;
    ov.z = (v[2] - mean) * inv * gv.z + bv.z;
    ov.w = (v[3] - mean) * inv * gv.w + bv.w;
    ((float4*)(out + (size_t)row * D_MODEL))[t] = ov;
}

// ----------------------------- launch --------------------------------------
extern "C" void kernel_launch(void* const* d_in, const int* in_sizes, int n_in,
                              void* d_out, int out_size, void* d_ws, size_t ws_size,
                              hipStream_t stream)
{
    const float* x   = (const float*)d_in[0];
    const int* esrc  = (const int*)  d_in[3];
    const int* edst  = (const int*)  d_in[4];
    const float* Wq  = (const float*)d_in[5];  const float* bq = (const float*)d_in[6];
    const float* Wk  = (const float*)d_in[7];  const float* bk = (const float*)d_in[8];
    const float* Wv  = (const float*)d_in[9];  const float* bv = (const float*)d_in[10];
    const float* Wo  = (const float*)d_in[11]; const float* bo = (const float*)d_in[12];
    const float* W1  = (const float*)d_in[13]; const float* b1 = (const float*)d_in[14];
    const float* W2  = (const float*)d_in[15]; const float* b2 = (const float*)d_in[16];
    const float* g1  = (const float*)d_in[17]; const float* be1 = (const float*)d_in[18];
    const float* g2  = (const float*)d_in[19]; const float* be2 = (const float*)d_in[20];

    const int N   = in_sizes[0] / D_MODEL;   // 10000
    const int E   = in_sizes[3];             // 160000
    const int DFF = in_sizes[14];            // 2048

    // workspace layout (floats): Q | K | V | wv | z | t
    float* ws = (float*)d_ws;
    const size_t nd = (size_t)N * D_MODEL;
    float* Qb  = ws;
    float* Kb  = ws + nd;
    float* Vb  = ws + 2 * nd;
    float* WVb = ws + 3 * nd;
    float* Zb  = ws + 4 * nd;                 // N*NHEAD
    float* Tb  = Zb + (size_t)N * NHEAD;      // N*DFF

    // zero the scatter accumulators (wv and z are contiguous)
    {
        const long cnt = (long)nd + (long)N * NHEAD;
        zero_f32<<<dim3((unsigned)((cnt + 255) / 256)), dim3(256), 0, stream>>>(WVb, cnt);
    }

    const dim3 gb(256);
    const dim3 gqkv((N + 63) / 64, D_MODEL / 128);

    // Q/K/V projections
    gemm_bias_act<<<gqkv, gb, 0, stream>>>(x, Wq, bq, Qb, N, D_MODEL, D_MODEL, 0);
    gemm_bias_act<<<gqkv, gb, 0, stream>>>(x, Wk, bk, Kb, N, D_MODEL, D_MODEL, 0);
    gemm_bias_act<<<gqkv, gb, 0, stream>>>(x, Wv, bv, Vb, N, D_MODEL, D_MODEL, 0);

    // edge attention scatter
    edge_attn<<<dim3((E + 7) / 8), dim3(256), 0, stream>>>(Qb, Kb, Vb, esrc, edst, WVb, Zb, E);

    // o = wv / z  (in place in WVb), float4-granular
    finalize_o<<<dim3((unsigned)((nd / 4 + 255) / 256)), dim3(256), 0, stream>>>(WVb, Zb, WVb, (int)(nd / 4));

    // output projection + LN1 (P -> Qb, h1 -> Kb)
    gemm_bias_act<<<gqkv, gb, 0, stream>>>(WVb, Wo, bo, Qb, N, D_MODEL, D_MODEL, 0);
    add_layernorm<<<dim3(N), dim3(128), 0, stream>>>(x, Qb, g1, be1, Kb);

    // FFN: t = relu(h1 @ W1 + b1); u = t @ W2 + b2 -> Qb; h2 = LN(h1 + u) -> out
    gemm_bias_act<<<dim3((N + 63) / 64, DFF / 128), gb, 0, stream>>>(Kb, W1, b1, Tb, N, D_MODEL, DFF, 1);
    gemm_bias_act<<<gqkv, gb, 0, stream>>>(Tb, W2, b2, Qb, N, DFF, D_MODEL, 0);
    add_layernorm<<<dim3(N), dim3(128), 0, stream>>>(Kb, Qb, g2, be2, (float*)d_out);
}
